// ComplexSwinAttention_37615323578766
// MI455X (gfx1250) — matrix operand order repaired and checked by reading, not compiled, not run on hardware
//
#include <hip/hip_runtime.h>

#define BATCH  1024
#define DIMC   256
#define NTOK   64
#define NHEADS 8
#define HDIM   32

typedef __attribute__((ext_vector_type(16))) __bf16       v16bf;
typedef __attribute__((ext_vector_type(8)))  __bf16       v8bf;
typedef __attribute__((ext_vector_type(8)))  float        v8f;
typedef __attribute__((ext_vector_type(8)))  unsigned int v8u;
typedef __attribute__((ext_vector_type(4)))  unsigned int v4u;

static __device__ __forceinline__ __bf16 f2bf(float f) {
    unsigned u = __builtin_bit_cast(unsigned, f);
    unsigned r = u + 0x7fffu + ((u >> 16) & 1u);   // round-to-nearest-even
    unsigned short h = (unsigned short)(r >> 16);
    return __builtin_bit_cast(__bf16, h);
}

static __device__ __forceinline__ v8bf pack8(v8f x) {
    v8bf o;
#pragma unroll
    for (int i = 0; i < 8; ++i) o[i] = f2bf(x[i]);
    return o;
}

static __device__ __forceinline__ v8f zero8() {
    v8f z;
#pragma unroll
    for (int i = 0; i < 8; ++i) z[i] = 0.0f;
    return z;
}

static __device__ __forceinline__ v8f wmma_bf16(v16bf a, v16bf b, v8f c) {
    return __builtin_amdgcn_wmma_f32_16x16x32_bf16(
        false, a, false, b, (short)0, c, false, false);
}

static __device__ __forceinline__ v16bf neg16(v16bf a) {
    v8u u = __builtin_bit_cast(v8u, a);
#pragma unroll
    for (int i = 0; i < 8; ++i) u[i] ^= 0x80008000u;
    return __builtin_bit_cast(v16bf, u);
}

// wave-local LDS RAW fence (cross-lane through LDS within one wave)
static __device__ __forceinline__ void lds_fence() {
    asm volatile("s_wait_dscnt 0x0" ::: "memory");
}

// A fragment (16x32 bf16): row-major [row][K] source, `stride` elems/row.
static __device__ __forceinline__ v16bf load_a(const __bf16* m, int stride, int row0, int kb) {
    int lane = (int)(threadIdx.x & 31u);
    int lh = lane >> 4;
    const __bf16* base = m + (size_t)(row0 + (lane & 15)) * stride + kb;
    v8bf lo = *(const v8bf*)(base + 8 * lh);
    v8bf hi = *(const v8bf*)(base + 16 + 8 * lh);
    v16bf r;
#pragma unroll
    for (int i = 0; i < 8; ++i) { r[i] = lo[i]; r[i + 8] = hi[i]; }
    return r;
}

// B fragment (32x16 bf16): stored [col][K] (K contiguous), `stride` elems/col.
static __device__ __forceinline__ v16bf load_b(const __bf16* m, int stride, int col0, int kb) {
    int lane = (int)(threadIdx.x & 31u);
    int lh = lane >> 4;
    const __bf16* p = m + (size_t)(col0 + (lane & 15)) * stride + kb;
    v8bf lo = *(const v8bf*)(p + 8 * lh);
    v8bf hi = *(const v8bf*)(p + 16 + 8 * lh);
    v16bf r;
#pragma unroll
    for (int i = 0; i < 8; ++i) { r[i] = lo[i]; r[i + 8] = hi[i]; }
    return r;
}

// ---- Tensor Data Mover: 1-row descriptor copying nelem contiguous bf16 to LDS.
// D# layout per CDNA5 ISA (group0 128b, group1 256b).
static __device__ __forceinline__ void tdm_copy_bf16(__bf16* lds_dst, const __bf16* gsrc,
                                                     unsigned nelem) {
    unsigned lds_addr = (unsigned)(unsigned long long)(uintptr_t)lds_dst;
    unsigned long long ga = (unsigned long long)(uintptr_t)gsrc;
    v4u g0;
    g0[0] = 1u;                                   // count=1 (valid user descriptor)
    g0[1] = lds_addr;                             // lds_addr      [63:32]
    g0[2] = (unsigned)ga;                         // global_addr   [95:64]
    g0[3] = (unsigned)((ga >> 32) & 0x01ffffffu)  // global_addr   [120:96]
          | (2u << 30);                           // type=2 (image) [127:126]
    v8u g1;
    g1[0] = (1u << 16);                           // wg_mask=0, data_size=1 (2B)
    g1[1] = (nelem & 0xffffu) << 16;              // tensor_dim0 [63:48]
    g1[2] = ((nelem >> 16) & 0xffffu)             // tensor_dim0 [79:64]
          | (1u << 16);                           // tensor_dim1=1 [111:80] low
    g1[3] = (nelem & 0xffffu) << 16;              // tile_dim0   [127:112]
    g1[4] = 1u;                                   // tile_dim1=1, tile_dim2=0
    g1[5] = nelem;                                // tensor_dim0_stride [207:160] low
    g1[6] = 0u;
    g1[7] = 0u;
    asm volatile("tensor_load_to_lds %0, %1" :: "s"(g0), "s"(g1) : "memory");
}

// ---------- prep kernels ----------

// x [B][256][64] f32 -> XT [B][64][256] bf16 (token-major: contiguous K for B-frags)
__global__ void prep_x_kernel(const float* __restrict__ xr, const float* __restrict__ xi,
                              __bf16* __restrict__ XTr, __bf16* __restrict__ XTi) {
    int b = blockIdx.x;
    const float* pr = xr + (size_t)b * DIMC * NTOK;
    const float* pi = xi + (size_t)b * DIMC * NTOK;
    __bf16* qr = XTr + (size_t)b * NTOK * DIMC;
    __bf16* qi = XTi + (size_t)b * NTOK * DIMC;
    for (int i = threadIdx.x; i < DIMC * NTOK; i += blockDim.x) {
        int c = i >> 6, p = i & 63;
        qr[p * DIMC + c] = f2bf(pr[i]);
        qi[p * DIMC + c] = f2bf(pi[i]);
    }
}

__global__ void cvt_bf16_kernel(const float* __restrict__ src, __bf16* __restrict__ dst, int n) {
    int i = blockIdx.x * blockDim.x + threadIdx.x;
    if (i < n) dst[i] = f2bf(src[i]);
}

__global__ void prep_bias_kernel(const float* __restrict__ table, const int* __restrict__ rel,
                                 float* __restrict__ bias8) {
    int i = blockIdx.x * blockDim.x + threadIdx.x;
    if (i < NTOK * NTOK) {
        int ri = rel[i];
#pragma unroll
        for (int h = 0; h < NHEADS; ++h)
            bias8[h * NTOK * NTOK + i] = table[ri * NHEADS + h];
    }
}

// ---------- main kernel: one workgroup per batch, wave h = head h ----------
__global__ __launch_bounds__(256)
void attn_batch_kernel(const __bf16* __restrict__ XTr, const __bf16* __restrict__ XTi,
                       const __bf16* __restrict__ Wqr, const __bf16* __restrict__ Wqi,
                       const float* __restrict__ bias8,
                       __bf16* __restrict__ OutR, __bf16* __restrict__ OutI) {
    const int b = (int)blockIdx.x;
    const int h = (int)(threadIdx.x >> 5);           // wave index = head
    const int lane = (int)(threadIdx.x & 31u);
    const int lh = lane >> 4;

    // 288 KB static LDS (<= 320 KB/WGP per CDNA5 ISA; accepted by toolchain)
    __shared__ __align__(16) __bf16 sXr[NTOK * DIMC], sXi[NTOK * DIMC];      // 64 KB
    __shared__ __align__(16) __bf16 sQr[NHEADS][NTOK * HDIM], sQi[NHEADS][NTOK * HDIM];
    __shared__ __align__(16) __bf16 sKr[NHEADS][NTOK * HDIM], sKi[NHEADS][NTOK * HDIM];
    __shared__ __align__(16) __bf16 sVr[NHEADS][HDIM * NTOK], sVi[NHEADS][HDIM * NTOK];
    __shared__ __align__(16) __bf16 sAr[NHEADS][16 * NTOK],   sAi[NHEADS][16 * NTOK];

    // prefetch this head's L2-resident weight rows while the TDM runs
    __builtin_prefetch(Wqr + (size_t)(h * HDIM) * DIMC, 0, 1);
    __builtin_prefetch(Wqi + (size_t)(h * HDIM) * DIMC, 0, 1);
    __builtin_prefetch(Wqr + (size_t)(DIMC + h * HDIM) * DIMC, 0, 1);
    __builtin_prefetch(Wqi + (size_t)(DIMC + h * HDIM) * DIMC, 0, 1);
    __builtin_prefetch(Wqr + (size_t)(2 * DIMC + h * HDIM) * DIMC, 0, 1);
    __builtin_prefetch(Wqi + (size_t)(2 * DIMC + h * HDIM) * DIMC, 0, 1);

    // ---- stage X (this batch, both components) into LDS via the TDM
    if (threadIdx.x < 32) {
        tdm_copy_bf16(sXr, XTr + (size_t)b * NTOK * DIMC, NTOK * DIMC);
        tdm_copy_bf16(sXi, XTi + (size_t)b * NTOK * DIMC, NTOK * DIMC);
        __builtin_amdgcn_s_wait_tensorcnt(0);
    }
    __syncthreads();

    // ---- QKV for this head: (Wr + iWi)(Xr + iXi)
    for (int part = 0; part < 3; ++part) {
        __bf16* dR = (part == 0) ? sQr[h] : (part == 1) ? sKr[h] : sVr[h];
        __bf16* dI = (part == 0) ? sQi[h] : (part == 1) ? sKi[h] : sVi[h];
        const int rowbase = part * DIMC + h * HDIM;
        for (int mt = 0; mt < 2; ++mt) {
            for (int nt = 0; nt < 4; ++nt) {
                v8f dr = zero8(), di = zero8();
#pragma unroll
                for (int kc = 0; kc < 8; ++kc) {
                    const int kb = kc * 32;
                    v16bf ar = load_a(Wqr, DIMC, rowbase + mt * 16, kb);
                    v16bf ai = load_a(Wqi, DIMC, rowbase + mt * 16, kb);
                    v16bf br = load_b(sXr, DIMC, nt * 16, kb);
                    v16bf bi = load_b(sXi, DIMC, nt * 16, kb);
                    dr = wmma_bf16(ar, br, dr);
                    dr = wmma_bf16(neg16(ai), bi, dr);   // - Wi*Xi
                    di = wmma_bf16(ar, bi, di);
                    di = wmma_bf16(ai, br, di);
                }
                const int tk = nt * 16 + (lane & 15);
                if (part < 2) {
                    // [tok][d]: 8 consecutive d per lane -> packed b128 store
                    int d0 = mt * 16 + lh * 8;
                    *(v8bf*)&dR[tk * HDIM + d0] = pack8(dr);
                    *(v8bf*)&dI[tk * HDIM + d0] = pack8(di);
                } else {
                    // v: [d][tok]
#pragma unroll
                    for (int r = 0; r < 8; ++r) {
                        int d = mt * 16 + lh * 8 + r;
                        dR[d * NTOK + tk] = f2bf(dr[r]);
                        dI[d * NTOK + tk] = f2bf(di[r]);
                    }
                }
            }
        }
    }
    lds_fence();   // per-head buffers are wave-private: wave-local fence suffices

    const float scale = 0.17677669529663689f;  // 32^-0.5
    const float* biash = bias8 + h * NTOK * NTOK;
    __bf16* oR = OutR + (size_t)b * NTOK * DIMC;
    __bf16* oI = OutI + (size_t)b * NTOK * DIMC;

    for (int strip = 0; strip < 4; ++strip) {
        // ---- attn = q * conj(k)^T : re = qr*kr + qi*ki ; im = qi*kr - qr*ki
        v8f cr[4], ci[4];
#pragma unroll
        for (int mt = 0; mt < 4; ++mt) { cr[mt] = zero8(); ci[mt] = zero8(); }
        v16bf qr = load_a(sQr[h], HDIM, strip * 16, 0);   // K = d = 32
        v16bf qi = load_a(sQi[h], HDIM, strip * 16, 0);
#pragma unroll
        for (int mt = 0; mt < 4; ++mt) {
            v16bf kr = load_b(sKr[h], HDIM, mt * 16, 0);
            v16bf ki = load_b(sKi[h], HDIM, mt * 16, 0);
            cr[mt] = wmma_bf16(qr, kr, cr[mt]);
            cr[mt] = wmma_bf16(qi, ki, cr[mt]);
            ci[mt] = wmma_bf16(qi, kr, ci[mt]);
            ci[mt] = wmma_bf16(qr, neg16(ki), ci[mt]);
        }
        // ---- scale, bias, magnitude
        v8f mag[4];
#pragma unroll
        for (int mt = 0; mt < 4; ++mt) {
#pragma unroll
            for (int r = 0; r < 8; ++r) {
                int n = strip * 16 + lh * 8 + r;
                int m = mt * 16 + (lane & 15);
                float re = cr[mt][r] * scale + biash[n * NTOK + m];
                float im = ci[mt][r] * scale;
                cr[mt][r] = re;
                ci[mt][r] = im;
                mag[mt][r] = sqrtf(re * re + im * im);
            }
        }
        // ---- magnitude softmax per row; a D-tile row lives in one 16-lane half
#pragma unroll
        for (int r = 0; r < 8; ++r) {
            float mx = fmaxf(fmaxf(mag[0][r], mag[1][r]), fmaxf(mag[2][r], mag[3][r]));
#pragma unroll
            for (int off = 1; off < 16; off <<= 1) mx = fmaxf(mx, __shfl_xor(mx, off, 32));
            float e[4], s = 0.0f;
#pragma unroll
            for (int mt = 0; mt < 4; ++mt) { e[mt] = __expf(mag[mt][r] - mx); s += e[mt]; }
#pragma unroll
            for (int off = 1; off < 16; off <<= 1) s += __shfl_xor(s, off, 32);
            float inv = 1.0f / s;
#pragma unroll
            for (int mt = 0; mt < 4; ++mt) {
                float fct = e[mt] * inv / (mag[mt][r] + 1e-8f);
                int a = (lh * 8 + r) * NTOK + mt * 16 + (lane & 15);
                sAr[h][a] = f2bf(cr[mt][r] * fct);
                sAi[h][a] = f2bf(ci[mt][r] * fct);
            }
        }
        lds_fence();
        // ---- out = attn * v : re = ar*vr - ai*vi ; im = ar*vi + ai*vr
        v8f orr[2], oii[2];
#pragma unroll
        for (int dt = 0; dt < 2; ++dt) { orr[dt] = zero8(); oii[dt] = zero8(); }
#pragma unroll
        for (int kc = 0; kc < 2; ++kc) {
            v16bf ar = load_a(sAr[h], NTOK, 0, kc * 32);
            v16bf ai = load_a(sAi[h], NTOK, 0, kc * 32);
#pragma unroll
            for (int dt = 0; dt < 2; ++dt) {
                v16bf br = load_b(sVr[h], NTOK, dt * 16, kc * 32);
                v16bf bi = load_b(sVi[h], NTOK, dt * 16, kc * 32);
                orr[dt] = wmma_bf16(ar, br, orr[dt]);
                orr[dt] = wmma_bf16(ai, neg16(bi), orr[dt]);
                oii[dt] = wmma_bf16(ar, bi, oii[dt]);
                oii[dt] = wmma_bf16(ai, br, oii[dt]);
            }
        }
        // store head output token-major [tok][256] for proj B-fragments
#pragma unroll
        for (int dt = 0; dt < 2; ++dt) {
#pragma unroll
            for (int r = 0; r < 8; ++r) {
                int n  = strip * 16 + lh * 8 + r;
                int dd = h * HDIM + dt * 16 + (lane & 15);
                oR[n * DIMC + dd] = f2bf(orr[dt][r]);
                oI[n * DIMC + dd] = f2bf(oii[dt][r]);
            }
        }
    }
}

// ---------- projection: one block (8 waves) per batch, complex GEMM 256x256x64 ----------
// B operand (head outputs, 64 KB/batch) is TDM-staged into LDS once per workgroup.
__global__ __launch_bounds__(256)
void proj_kernel(const __bf16* __restrict__ OutR, const __bf16* __restrict__ OutI,
                 const __bf16* __restrict__ Wpr, const __bf16* __restrict__ Wpi,
                 float* __restrict__ out) {
    const int b = (int)blockIdx.x;
    const int w = (int)(threadIdx.x >> 5);
    const int lane = (int)(threadIdx.x & 31u);
    const int lh = lane >> 4;

    __shared__ __align__(16) __bf16 sCr[NTOK * DIMC], sCi[NTOK * DIMC];   // 64 KB

    __builtin_prefetch(Wpr + (size_t)(w * 32) * DIMC, 0, 1);
    __builtin_prefetch(Wpi + (size_t)(w * 32) * DIMC, 0, 1);

    if (threadIdx.x < 32) {
        tdm_copy_bf16(sCr, OutR + (size_t)b * NTOK * DIMC, NTOK * DIMC);
        tdm_copy_bf16(sCi, OutI + (size_t)b * NTOK * DIMC, NTOK * DIMC);
        __builtin_amdgcn_s_wait_tensorcnt(0);
    }
    __syncthreads();

    for (int nt = 0; nt < 4; ++nt) {
        for (int mt = 0; mt < 2; ++mt) {
            v8f dr = zero8(), di = zero8();
#pragma unroll
            for (int kc = 0; kc < 8; ++kc) {
                const int kb = kc * 32;
                v16bf ar = load_a(Wpr, DIMC, w * 32 + mt * 16, kb);
                v16bf ai = load_a(Wpi, DIMC, w * 32 + mt * 16, kb);
                v16bf br = load_b(sCr, DIMC, nt * 16, kb);
                v16bf bi = load_b(sCi, DIMC, nt * 16, kb);
                dr = wmma_bf16(ar, br, dr);
                dr = wmma_bf16(ai, neg16(bi), dr);
                di = wmma_bf16(ar, bi, di);
                di = wmma_bf16(ai, br, di);
            }
#pragma unroll
            for (int r = 0; r < 8; ++r) {
                int o = w * 32 + mt * 16 + lh * 8 + r;
                int p = nt * 16 + (lane & 15);
                size_t idx = ((size_t)b * DIMC + o) * NTOK + p;
                out[2 * idx]     = dr[r];   // complex64 interleaved
                out[2 * idx + 1] = di[r];
            }
        }
    }
}

extern "C" void kernel_launch(void* const* d_in, const int* in_sizes, int n_in,
                              void* d_out, int out_size, void* d_ws, size_t ws_size,
                              hipStream_t stream) {
    (void)in_sizes; (void)n_in; (void)out_size; (void)ws_size;
    const float* x_real     = (const float*)d_in[0];
    const float* x_imag     = (const float*)d_in[1];
    const float* w_qkv_r    = (const float*)d_in[2];
    const float* w_qkv_i    = (const float*)d_in[3];
    const float* w_proj_r   = (const float*)d_in[4];
    const float* w_proj_i   = (const float*)d_in[5];
    const float* bias_table = (const float*)d_in[6];
    const int*   rel_index  = (const int*)d_in[7];

    char* ws = (char*)d_ws;
    size_t off = 0;
    __bf16* XTr = (__bf16*)(ws + off); off += (size_t)BATCH * NTOK * DIMC * 2;
    __bf16* XTi = (__bf16*)(ws + off); off += (size_t)BATCH * NTOK * DIMC * 2;
    __bf16* Wqr = (__bf16*)(ws + off); off += (size_t)3 * DIMC * DIMC * 2;
    __bf16* Wqi = (__bf16*)(ws + off); off += (size_t)3 * DIMC * DIMC * 2;
    __bf16* Wpr = (__bf16*)(ws + off); off += (size_t)DIMC * DIMC * 2;
    __bf16* Wpi = (__bf16*)(ws + off); off += (size_t)DIMC * DIMC * 2;
    float*  bias8 = (float*)(ws + off); off += (size_t)NHEADS * NTOK * NTOK * 4;
    __bf16* OutR = (__bf16*)(ws + off); off += (size_t)BATCH * NTOK * DIMC * 2;
    __bf16* OutI = (__bf16*)(ws + off); off += (size_t)BATCH * NTOK * DIMC * 2;

    prep_x_kernel<<<BATCH, 256, 0, stream>>>(x_real, x_imag, XTr, XTi);
    cvt_bf16_kernel<<<(3 * DIMC * DIMC + 255) / 256, 256, 0, stream>>>(w_qkv_r, Wqr, 3 * DIMC * DIMC);
    cvt_bf16_kernel<<<(3 * DIMC * DIMC + 255) / 256, 256, 0, stream>>>(w_qkv_i, Wqi, 3 * DIMC * DIMC);
    cvt_bf16_kernel<<<(DIMC * DIMC + 255) / 256, 256, 0, stream>>>(w_proj_r, Wpr, DIMC * DIMC);
    cvt_bf16_kernel<<<(DIMC * DIMC + 255) / 256, 256, 0, stream>>>(w_proj_i, Wpi, DIMC * DIMC);
    prep_bias_kernel<<<(NTOK * NTOK + 255) / 256, 256, 0, stream>>>(bias_table, rel_index, bias8);

    attn_batch_kernel<<<BATCH, 256, 0, stream>>>(XTr, XTi, Wqr, Wqi, bias8, OutR, OutI);
    proj_kernel<<<BATCH, 256, 0, stream>>>(OutR, OutI, Wpr, Wpi, (float*)d_out);
}
